// TrajDecoder_32212254720757
// MI455X (gfx1250) — matrix operand order repaired and checked by reading, not compiled, not run on hardware
//
#include <hip/hip_runtime.h>
#include <hip/hip_bf16.h>
#include <math.h>

// ---------------- problem constants (from the reference) ----------------
#define N_T   102400
#define N_L   8192
#define N_M   12288
#define E_T2M 122880
#define E_L2M 196608
#define E_M2M 393216
#define DIM   128
#define NHEAD 8
#define THIST 50
#define FOUT  120          // F*2

typedef __attribute__((ext_vector_type(16))) __bf16 v16bf;
typedef __attribute__((ext_vector_type(8)))  __bf16 v8bf;
typedef __attribute__((ext_vector_type(8)))  float  v8f;
typedef __attribute__((ext_vector_type(4)))  float  fx4;   // builtin vec (AS-cast friendly)

// global (address_space(1)) pointer helpers to force global_load_* paths
typedef __attribute__((address_space(1))) const fx4   gcf4;
typedef __attribute__((address_space(1))) const v8bf  gcv8bf;
typedef __attribute__((address_space(1))) const v16bf gcv16;

// ---------------- weight repack: fp32 [Kact,Nact] -> bf16 WMMA B-fragments ----
// Fragment f = kt*(Npad/16) + nt ; per lane 16 contiguous bf16:
//   Wp[(f*32 + lane)*16 + i] = W[kt*32 + (i>>3)*16 + (lane>>4)*8 + (i&7) , nt*16 + (lane&15)]
__global__ void pack_w_kernel(const float* W, __bf16* Wp,
                              int Kact, int Nact, int Kpad, int Npad) {
  const int nnt = Npad >> 4;
  const int total = (Kpad >> 5) * nnt * 32;
  int t = blockIdx.x * blockDim.x + threadIdx.x;
  if (t >= total) return;
  const int lane = t & 31;
  const int f = t >> 5;
  const int nt = f % nnt;
  const int kt = f / nnt;
  const int n = nt * 16 + (lane & 15);
  const int kb = kt * 32 + (lane >> 4) * 8;
  __bf16* dst = Wp + (size_t)t * 16;
  #pragma unroll
  for (int i = 0; i < 16; ++i) {
    const int k = kb + (i >> 3) * 16 + (i & 7);
    float v = (k < Kact && n < Nact) ? W[(size_t)k * Nact + n] : 0.f;
    dst[i] = (__bf16)v;
  }
}

// ---------------- WMMA GEMM: out[M,N] = act(concat-A @ W + bias) + add ----
// A = concat of up to 3 segments, each optionally row-gathered, fp32 or bf16.
struct GemmSeg { const void* A; const int* idx; int ld; int width; int isbf; };
struct GemmArgs {
  GemmSeg seg[3];
  int nseg;
  const __bf16* Wp; int Kpad; int Npad;   // packed bf16 fragments
  const float* bias;
  const float* add; const int* addidx; int ldadd;
  void* out; int ldout; int outbf;
  int M, Nact, relu;
};

// block = 256 threads = 8 waves; block tile = 64(M) x 128(N);
// wave tile = 16(M) x 64(N): mt = wave>>1, ntile base = (wave&1)*4.
__global__ __launch_bounds__(256) void wmma_gemm_kernel(GemmArgs g) {
  const int lane  = threadIdx.x & 31;
  const int wave  = threadIdx.x >> 5;
  const int mt    = wave >> 1;
  const int ng    = (wave & 1) << 2;
  const int halfw = lane >> 4;
  const int l16   = lane & 15;
  const int m0 = blockIdx.x * 64 + mt * 16;
  int rm = m0 + l16; if (rm > g.M - 1) rm = g.M - 1;

  const int nnt = g.Npad >> 4;      // fragments per k-tile row (8 when Npad==128)
  const int fstep = nnt * 32;       // v16bf elements per k-tile of fragments
  gcv16* wptr = (gcv16*)g.Wp + (size_t)ng * 32 + lane;

  v8f acc[4];
  #pragma unroll
  for (int j = 0; j < 4; ++j) acc[j] = (v8f){0.f,0.f,0.f,0.f,0.f,0.f,0.f,0.f};

  // loop over concat segments; per segment a tight inner loop with hoisted
  // row pointer / dtype (nseg==1 for most GEMMs -> single clean loop)
  for (int s = 0; s < g.nseg; ++s) {
    const GemmSeg sg = g.seg[s];
    const int r = sg.idx ? sg.idx[rm] : rm;
    const int nk = sg.width >> 5;   // k-tiles in this segment
    if (sg.isbf) {
      // A rows in bf16: lane reads two 16B runs (8 bf16 each), no conversion
      const char* rp = (const char*)sg.A + ((size_t)r * (size_t)sg.ld) * 2 + halfw * 16;
      for (int t = 0; t < nk; ++t) {
        const v8bf lo = *(gcv8bf*)(rp);
        const v8bf hi = *(gcv8bf*)(rp + 32);
        __builtin_prefetch(rp + 64, 0, 3);
        v16bf a;
        #pragma unroll
        for (int i = 0; i < 8; ++i) { a[i] = lo[i]; a[8 + i] = hi[i]; }
        #pragma unroll
        for (int j = 0; j < 4; ++j) {
          const v16bf b = wptr[(size_t)j * 32];
          acc[j] = __builtin_amdgcn_wmma_f32_16x16x32_bf16(
              false, a, false, b, (short)0, acc[j], false, false);
        }
        rp += 64;          // 32 bf16 elements
        wptr += fstep;
      }
    } else {
      // A rows in fp32: two 2xb128 runs + cvt_pk_bf16
      const float* rf = (const float*)sg.A + (size_t)r * (size_t)sg.ld + halfw * 8;
      for (int t = 0; t < nk; ++t) {
        gcf4* pa = (gcf4*)(rf);
        gcf4* pb = (gcf4*)(rf + 16);
        const fx4 a0 = pa[0], a1 = pa[1];
        const fx4 a2 = pb[0], a3 = pb[1];
        __builtin_prefetch(rf + 32, 0, 3);
        v16bf a;
        a[0]=(__bf16)a0[0]; a[1]=(__bf16)a0[1]; a[2]=(__bf16)a0[2]; a[3]=(__bf16)a0[3];
        a[4]=(__bf16)a1[0]; a[5]=(__bf16)a1[1]; a[6]=(__bf16)a1[2]; a[7]=(__bf16)a1[3];
        a[8]=(__bf16)a2[0]; a[9]=(__bf16)a2[1]; a[10]=(__bf16)a2[2]; a[11]=(__bf16)a2[3];
        a[12]=(__bf16)a3[0]; a[13]=(__bf16)a3[1]; a[14]=(__bf16)a3[2]; a[15]=(__bf16)a3[3];
        #pragma unroll
        for (int j = 0; j < 4; ++j) {
          const v16bf b = wptr[(size_t)j * 32];
          acc[j] = __builtin_amdgcn_wmma_f32_16x16x32_bf16(
              false, a, false, b, (short)0, acc[j], false, false);
        }
        rf += 32;
        wptr += fstep;
      }
    }
  }

  // C/D layout: lane n = lane&15 ; VGPR r -> m = r + 8*(lane>>4)
  #pragma unroll
  for (int j = 0; j < 4; ++j) {
    const int ncol = (ng + j) * 16 + l16;
    float bv = 0.f;
    if (g.bias && ncol < g.Nact) bv = g.bias[ncol];
    #pragma unroll
    for (int r = 0; r < 8; ++r) {
      const int m = m0 + r + 8 * halfw;
      if (m < g.M && ncol < g.Nact) {
        float c = acc[j][r] + bv;
        if (g.relu) c = fmaxf(c, 0.f);
        if (g.add) {
          const int rr = g.addidx ? g.addidx[m] : m;
          c += g.add[(size_t)rr * (size_t)g.ldadd + ncol];
        }
        if (g.outbf) ((__bf16*)g.out)[(size_t)m * (size_t)g.ldout + ncol] = (__bf16)c;
        else         ((float*)g.out)[(size_t)m * (size_t)g.ldout + ncol] = c;
      }
    }
  }
}

// ---------------- element-wise kernels ----------------
__global__ void init_modes_kernel(const float* tok, float* m, int total) {
  int i = blockIdx.x * blockDim.x + threadIdx.x;
  if (i >= total) return;
  int row = i >> 7, col = i & 127;
  m[i] = tok[(row % 6) * DIM + col];
}

__global__ void edge_feat_kernel(const float* spos, const float* shead,
                                 const float* dpos, const float* dhead,
                                 const int* src, const int* dst, int E,
                                 int withInterval, float* feat /* [E,32] */) {
  int e = blockIdx.x * blockDim.x + threadIdx.x;
  if (e >= E) return;
  const int s = src[e], d = dst[e];
  const float rx = spos[2 * s] - dpos[2 * d];
  const float ry = spos[2 * s + 1] - dpos[2 * d + 1];
  const float hd = dhead[d];
  const float c = cosf(hd), sn = sinf(hd);
  const float vx = c * rx + sn * ry;
  const float vy = -sn * rx + c * ry;
  const float len = sqrtf(vx * vx + vy * vy);
  const float th = atan2f(vy, vx);
  const float PI = 3.14159265358979323846f;
  float w = (shead[s] - hd) + PI;
  w = w - floorf(w * (0.5f / PI)) * (2.0f * PI);
  w -= PI;
  float* fp = feat + (size_t)e * 32;
  fp[0] = len; fp[1] = sinf(th); fp[2] = cosf(th); fp[3] = sinf(w); fp[4] = cosf(w);
  int i = 5;
  if (withInterval) { fp[5] = (float)(s % THIST - THIST); i = 6; }
  for (; i < 32; ++i) fp[i] = 0.f;
}

__global__ void logits_kernel(const float* qn, const float* kedge,
                              const int* dst, float* logits, int EH) {
  int t = blockIdx.x * blockDim.x + threadIdx.x;
  if (t >= EH) return;
  const int e = t >> 3, h = t & 7;
  const float* q = qn + (size_t)dst[e] * DIM + h * 16;
  const float* k = kedge + (size_t)e * DIM + h * 16;
  float s = 0.f;
  #pragma unroll
  for (int i = 0; i < 16; ++i) s += q[i] * k[i];
  logits[t] = s * 0.25f;   // 1/sqrt(16)
}

__global__ void init_softmax_kernel(float* rowmax, float* denom, int n) {
  int i = blockIdx.x * blockDim.x + threadIdx.x;
  if (i >= n) return;
  rowmax[i] = -1e30f;
  denom[i] = 0.f;
}

__device__ inline void atomicMaxF(float* addr, float val) {
  if (val >= 0.f) atomicMax((int*)addr, __float_as_int(val));
  else            atomicMin((unsigned int*)addr, __float_as_uint(val));
}

__global__ void segmax_kernel(const float* logits, const int* dst,
                              float* rowmax, int EH) {
  int t = blockIdx.x * blockDim.x + threadIdx.x;
  if (t >= EH) return;
  atomicMaxF(&rowmax[dst[t >> 3] * NHEAD + (t & 7)], logits[t]);
}

__global__ void segexp_kernel(float* logits, const int* dst,
                              const float* rowmax, float* denom, int EH) {
  int t = blockIdx.x * blockDim.x + threadIdx.x;
  if (t >= EH) return;
  const int rd = dst[t >> 3] * NHEAD + (t & 7);
  const float v = __expf(logits[t] - rowmax[rd]);
  logits[t] = v;
  atomicAdd(&denom[rd], v);
}

__global__ void segnorm_kernel(float* logits, const int* dst,
                               const float* denom, int EH) {
  int t = blockIdx.x * blockDim.x + threadIdx.x;
  if (t >= EH) return;
  logits[t] = logits[t] / (denom[dst[t >> 3] * NHEAD + (t & 7)] + 1e-9f);
}

__global__ void zero_kernel(float* p, int n) {
  int i = blockIdx.x * blockDim.x + threadIdx.x;
  if (i < n) p[i] = 0.f;
}

__global__ void agg_kernel(const float* alpha, const float* vedge,
                           const int* dst, float* agg, int E128) {
  int t = blockIdx.x * blockDim.x + threadIdx.x;
  if (t >= E128) return;
  const int e = t >> 7, dcol = t & 127, h = dcol >> 4;
  const float a = alpha[e * NHEAD + h];
  atomicAdd(&agg[(size_t)dst[e] * DIM + dcol], a * vedge[(size_t)e * DIM + dcol]);
}

// ---------------- host-side helpers ----------------
struct PK { const __bf16* p; int Kpad, Npad; };

static inline dim3 g1d(long long n, int b) { return dim3((unsigned)((n + b - 1) / b)); }

static inline void gemmN(hipStream_t st, GemmSeg s0, GemmSeg s1, GemmSeg s2, int nseg,
                         PK W, const float* bias,
                         const float* add, const int* addidx, int ldadd,
                         void* out, int ldout, int outbf, int M, int Nact, int relu) {
  GemmArgs g{};
  g.seg[0] = s0; g.seg[1] = s1; g.seg[2] = s2; g.nseg = nseg;
  g.Wp = W.p; g.Kpad = W.Kpad; g.Npad = W.Npad;
  g.bias = bias; g.add = add; g.addidx = addidx; g.ldadd = ldadd;
  g.out = out; g.ldout = ldout; g.outbf = outbf;
  g.M = M; g.Nact = Nact; g.relu = relu;
  dim3 grid((unsigned)((M + 63) / 64));
  wmma_gemm_kernel<<<grid, 256, 0, st>>>(g);
}

static inline void gemm1(hipStream_t st, const void* A, int isbf, const int* idx, int ld, int K,
                         PK W, const float* bias,
                         const float* add, const int* addidx, int ldadd,
                         void* out, int ldout, int outbf, int M, int Nact, int relu) {
  GemmSeg s0{A, idx, ld, K, isbf}, z{nullptr, nullptr, 0, 0, 0};
  gemmN(st, s0, z, z, 1, W, bias, add, addidx, ldadd, out, ldout, outbf, M, Nact, relu);
}

// one edge-conditioned multi-head graph attention block
static void run_attn(hipStream_t st,
                     const float* xsrc, int Nsrc, const float* xdst,
                     const int* src, const int* dst, int E, const __bf16* ebuf,
                     PK wq, PK wk, PK wv, PK wo, PK wke, PK wve,
                     float* qn, float* kn, float* vn, float* kedge, float* vedge,
                     float* logits, float* rowmax, float* denom, float* agg,
                     float* outdst) {
  gemm1(st, xdst, 0, nullptr, DIM, DIM, wq, nullptr, nullptr, nullptr, 0, qn, DIM, 0, N_M, DIM, 0);
  gemm1(st, xsrc, 0, nullptr, DIM, DIM, wk, nullptr, nullptr, nullptr, 0, kn, DIM, 0, Nsrc, DIM, 0);
  gemm1(st, xsrc, 0, nullptr, DIM, DIM, wv, nullptr, nullptr, nullptr, 0, vn, DIM, 0, Nsrc, DIM, 0);
  // k_edge = e @ wke + kn[src] ; v_edge = e @ wve + vn[src]
  gemm1(st, ebuf, 1, nullptr, DIM, DIM, wke, nullptr, kn, src, DIM, kedge, DIM, 0, E, DIM, 0);
  gemm1(st, ebuf, 1, nullptr, DIM, DIM, wve, nullptr, vn, src, DIM, vedge, DIM, 0, E, DIM, 0);
  const int EH = E * NHEAD;
  logits_kernel<<<g1d(EH, 256), 256, 0, st>>>(qn, kedge, dst, logits, EH);
  init_softmax_kernel<<<g1d(N_M * NHEAD, 256), 256, 0, st>>>(rowmax, denom, N_M * NHEAD);
  segmax_kernel<<<g1d(EH, 256), 256, 0, st>>>(logits, dst, rowmax, EH);
  segexp_kernel<<<g1d(EH, 256), 256, 0, st>>>(logits, dst, rowmax, denom, EH);
  segnorm_kernel<<<g1d(EH, 256), 256, 0, st>>>(logits, dst, denom, EH);
  zero_kernel<<<g1d(N_M * DIM, 256), 256, 0, st>>>(agg, N_M * DIM);
  agg_kernel<<<g1d((long long)E * DIM, 256), 256, 0, st>>>(logits, vedge, dst, agg, E * DIM);
  // out = x_dst + agg @ wo
  gemm1(st, agg, 0, nullptr, DIM, DIM, wo, nullptr, xdst, nullptr, DIM, outdst, DIM, 0, N_M, DIM, 0);
}

extern "C" void kernel_launch(void* const* d_in, const int* in_sizes, int n_in,
                              void* d_out, int out_size, void* d_ws, size_t ws_size,
                              hipStream_t stream) {
  // ---- inputs (setup_inputs dict order) ----
  const float* t_embs = (const float*)d_in[0];
  const float* l_embs = (const float*)d_in[1];
  const float* t_pos  = (const float*)d_in[2];
  const float* t_head = (const float*)d_in[3];
  const float* l_pos  = (const float*)d_in[4];
  const float* l_head = (const float*)d_in[5];
  const float* m_pos  = (const float*)d_in[6];
  const float* m_head = (const float*)d_in[7];
  const int* t2m_src = (const int*)d_in[8];
  const int* t2m_dst = (const int*)d_in[9];
  const int* l2m_src = (const int*)d_in[10];
  const int* l2m_dst = (const int*)d_in[11];
  const int* m2m_src = (const int*)d_in[12];
  const int* m2m_dst = (const int*)d_in[13];
  // ---- params (dict insertion order) ----
  int pi = 14;
  const float* mode_tokens = (const float*)d_in[pi++];
  const float* t2m_attr_w1 = (const float*)d_in[pi++];
  const float* t2m_attr_b1 = (const float*)d_in[pi++];
  const float* t2m_attr_w2 = (const float*)d_in[pi++];
  const float* t2m_attr_b2 = (const float*)d_in[pi++];
  const float* t2m_upd_w1  = (const float*)d_in[pi++];
  const float* t2m_upd_b1  = (const float*)d_in[pi++];
  const float* t2m_upd_w2  = (const float*)d_in[pi++];
  const float* t2m_upd_b2  = (const float*)d_in[pi++];
  const float* l2m_attr_w1 = (const float*)d_in[pi++];
  const float* l2m_attr_b1 = (const float*)d_in[pi++];
  const float* l2m_attr_w2 = (const float*)d_in[pi++];
  const float* l2m_attr_b2 = (const float*)d_in[pi++];
  const float* l2m_upd_w1  = (const float*)d_in[pi++];
  const float* l2m_upd_b1  = (const float*)d_in[pi++];
  const float* l2m_upd_w2  = (const float*)d_in[pi++];
  const float* l2m_upd_b2  = (const float*)d_in[pi++];
  const float* m2m_attr_w1 = (const float*)d_in[pi++];
  const float* m2m_attr_b1 = (const float*)d_in[pi++];
  const float* m2m_attr_w2 = (const float*)d_in[pi++];
  const float* m2m_attr_b2 = (const float*)d_in[pi++];
  const float* attn_w[4][6];  // t2m, l2m, m2m0, m2m1 : wq wk wv wo wke wve
  for (int a = 0; a < 4; ++a)
    for (int wj = 0; wj < 6; ++wj) attn_w[a][wj] = (const float*)d_in[pi++];
  const float* traj_w1 = (const float*)d_in[pi++];
  const float* traj_b1 = (const float*)d_in[pi++];
  const float* traj_w2 = (const float*)d_in[pi++];
  const float* traj_b2 = (const float*)d_in[pi++];

  // ---- workspace carve-up (256B aligned) ----
  char* w = (char*)d_ws;
  auto alloc = [&](size_t bytes) -> void* {
    void* p = (void*)w;
    w += (bytes + 255) & ~(size_t)255;
    return p;
  };

  // pack all weights to bf16 WMMA fragments (one-time, deterministic)
  auto packW = [&](const float* W, int Kact, int Nact) -> PK {
    const int Kpad = (Kact + 31) & ~31;
    const int Npad = (Nact + 15) & ~15;
    __bf16* dst = (__bf16*)alloc((size_t)Kpad * Npad * sizeof(__bf16));
    const int threads = (Kpad >> 5) * (Npad >> 4) * 32;
    pack_w_kernel<<<g1d(threads, 256), 256, 0, stream>>>(W, dst, Kact, Nact, Kpad, Npad);
    return {dst, Kpad, Npad};
  };

  PK p_t2m_a1 = packW(t2m_attr_w1, 6, DIM);
  PK p_t2m_a2 = packW(t2m_attr_w2, DIM, DIM);
  PK p_t2m_u1 = packW(t2m_upd_w1, 3 * DIM, DIM);
  PK p_t2m_u2 = packW(t2m_upd_w2, DIM, DIM);
  PK p_l2m_a1 = packW(l2m_attr_w1, 5, DIM);
  PK p_l2m_a2 = packW(l2m_attr_w2, DIM, DIM);
  PK p_l2m_u1 = packW(l2m_upd_w1, 3 * DIM, DIM);
  PK p_l2m_u2 = packW(l2m_upd_w2, DIM, DIM);
  PK p_m2m_a1 = packW(m2m_attr_w1, 5, DIM);
  PK p_m2m_a2 = packW(m2m_attr_w2, DIM, DIM);
  PK p_attn[4][6];
  for (int a = 0; a < 4; ++a)
    for (int wj = 0; wj < 6; ++wj) p_attn[a][wj] = packW(attn_w[a][wj], DIM, DIM);
  PK p_traj1 = packW(traj_w1, DIM, DIM);
  PK p_traj2 = packW(traj_w2, DIM, FOUT);

  const int EMAX = E_M2M;
  float*  mA     = (float*)alloc((size_t)N_M * DIM * 4);
  float*  mB     = (float*)alloc((size_t)N_M * DIM * 4);
  float*  qn     = (float*)alloc((size_t)N_M * DIM * 4);
  __bf16* qh     = (__bf16*)alloc((size_t)N_M * DIM * 2);  // bf16 traj hidden
  float*  kn     = (float*)alloc((size_t)N_T * DIM * 4);
  float*  vn     = (float*)alloc((size_t)N_T * DIM * 4);
  float*  feat   = (float*)alloc((size_t)EMAX * 32 * 4);
  __bf16* ebuf   = (__bf16*)alloc((size_t)EMAX * DIM * 2);  // edge attr (bf16)
  __bf16* hbuf   = (__bf16*)alloc((size_t)EMAX * DIM * 2);  // MLP hidden (bf16)
  float*  kedge  = (float*)alloc((size_t)EMAX * DIM * 4);
  float*  vedge  = (float*)alloc((size_t)EMAX * DIM * 4);
  float*  logits = (float*)alloc((size_t)EMAX * NHEAD * 4);
  float*  rowmax = (float*)alloc((size_t)N_M * NHEAD * 4);
  float*  denom  = (float*)alloc((size_t)N_M * NHEAD * 4);
  float*  agg    = (float*)alloc((size_t)N_M * DIM * 4);

  // ---- m_embs = tile(mode_tokens) ----
  init_modes_kernel<<<g1d(N_M * DIM, 256), 256, 0, stream>>>(mode_tokens, mA, N_M * DIM);

  // ================= t2m =================
  edge_feat_kernel<<<g1d(E_T2M, 256), 256, 0, stream>>>(
      t_pos, t_head, m_pos, m_head, t2m_src, t2m_dst, E_T2M, 1, feat);
  gemm1(stream, feat, 0, nullptr, 32, 32, p_t2m_a1, t2m_attr_b1,
        nullptr, nullptr, 0, hbuf, DIM, 1, E_T2M, DIM, 1);
  gemm1(stream, hbuf, 1, nullptr, DIM, DIM, p_t2m_a2, t2m_attr_b2,
        nullptr, nullptr, 0, ebuf, DIM, 1, E_T2M, DIM, 0);
  // upd MLP on concat([e(bf16), t_embs[src](f32), m_embs[dst](f32)])  (K = 384)
  gemmN(stream,
        GemmSeg{ebuf, nullptr, DIM, DIM, 1},
        GemmSeg{t_embs, t2m_src, DIM, DIM, 0},
        GemmSeg{mA, t2m_dst, DIM, DIM, 0}, 3,
        p_t2m_u1, t2m_upd_b1, nullptr, nullptr, 0, hbuf, DIM, 1, E_T2M, DIM, 1);
  gemm1(stream, hbuf, 1, nullptr, DIM, DIM, p_t2m_u2, t2m_upd_b2,
        nullptr, nullptr, 0, ebuf, DIM, 1, E_T2M, DIM, 0);
  run_attn(stream, t_embs, N_T, mA, t2m_src, t2m_dst, E_T2M, ebuf,
           p_attn[0][0], p_attn[0][1], p_attn[0][2], p_attn[0][3], p_attn[0][4], p_attn[0][5],
           qn, kn, vn, kedge, vedge, logits, rowmax, denom, agg, mB);

  // ================= l2m =================
  edge_feat_kernel<<<g1d(E_L2M, 256), 256, 0, stream>>>(
      l_pos, l_head, m_pos, m_head, l2m_src, l2m_dst, E_L2M, 0, feat);
  gemm1(stream, feat, 0, nullptr, 32, 32, p_l2m_a1, l2m_attr_b1,
        nullptr, nullptr, 0, hbuf, DIM, 1, E_L2M, DIM, 1);
  gemm1(stream, hbuf, 1, nullptr, DIM, DIM, p_l2m_a2, l2m_attr_b2,
        nullptr, nullptr, 0, ebuf, DIM, 1, E_L2M, DIM, 0);
  gemmN(stream,
        GemmSeg{ebuf, nullptr, DIM, DIM, 1},
        GemmSeg{l_embs, l2m_src, DIM, DIM, 0},
        GemmSeg{mB, l2m_dst, DIM, DIM, 0}, 3,
        p_l2m_u1, l2m_upd_b1, nullptr, nullptr, 0, hbuf, DIM, 1, E_L2M, DIM, 1);
  gemm1(stream, hbuf, 1, nullptr, DIM, DIM, p_l2m_u2, l2m_upd_b2,
        nullptr, nullptr, 0, ebuf, DIM, 1, E_L2M, DIM, 0);
  run_attn(stream, l_embs, N_L, mB, l2m_src, l2m_dst, E_L2M, ebuf,
           p_attn[1][0], p_attn[1][1], p_attn[1][2], p_attn[1][3], p_attn[1][4], p_attn[1][5],
           qn, kn, vn, kedge, vedge, logits, rowmax, denom, agg, mA);

  // ================= m2m (2 layers, shared edge attr) =================
  edge_feat_kernel<<<g1d(E_M2M, 256), 256, 0, stream>>>(
      m_pos, m_head, m_pos, m_head, m2m_src, m2m_dst, E_M2M, 0, feat);
  gemm1(stream, feat, 0, nullptr, 32, 32, p_m2m_a1, m2m_attr_b1,
        nullptr, nullptr, 0, hbuf, DIM, 1, E_M2M, DIM, 1);
  gemm1(stream, hbuf, 1, nullptr, DIM, DIM, p_m2m_a2, m2m_attr_b2,
        nullptr, nullptr, 0, ebuf, DIM, 1, E_M2M, DIM, 0);
  run_attn(stream, mA, N_M, mA, m2m_src, m2m_dst, E_M2M, ebuf,
           p_attn[2][0], p_attn[2][1], p_attn[2][2], p_attn[2][3], p_attn[2][4], p_attn[2][5],
           qn, kn, vn, kedge, vedge, logits, rowmax, denom, agg, mB);
  run_attn(stream, mB, N_M, mB, m2m_src, m2m_dst, E_M2M, ebuf,
           p_attn[3][0], p_attn[3][1], p_attn[3][2], p_attn[3][3], p_attn[3][4], p_attn[3][5],
           qn, kn, vn, kedge, vedge, logits, rowmax, denom, agg, mA);

  // ================= trajectory head =================
  gemm1(stream, mA, 0, nullptr, DIM, DIM, p_traj1, traj_b1,
        nullptr, nullptr, 0, qh, DIM, 1, N_M, DIM, 1);
  gemm1(stream, qh, 1, nullptr, DIM, DIM, p_traj2, traj_b2,
        nullptr, nullptr, 0, d_out, FOUT, 0, N_M, FOUT, 0);

  (void)in_sizes; (void)n_in; (void)out_size; (void)ws_size;
}
